// BoxHead_33277406609979
// MI455X (gfx1250) — matrix operand order repaired and checked by reading, not compiled, not run on hardware
//
#include <hip/hip_runtime.h>
#include <math.h>

// ---------------------------------------------------------------------------
// BoxHead on MI455X (gfx1250): f16 WMMA GEMMs with f32 accumulate,
// double-buffered async global->LDS tile staging (ASYNCcnt pipeline).
//   h1 = relu(X @ W1 + b1)        M=8000  K=12544 N=1024
//   h2 = relu(h1 @ W2 + b2)       M=8000  K=1024  N=1024
//   logits = h2 @ [Wc|Wr]         M=8000  K=1024  N=16
//   out = [softmax(logits[:, :4] + bc) ; logits[:, 4:] + br]
// ---------------------------------------------------------------------------

typedef __attribute__((ext_vector_type(16))) _Float16 v16h;
typedef __attribute__((ext_vector_type(8)))  _Float16 v8h;
typedef __attribute__((ext_vector_type(8)))  float    v8f;
typedef __attribute__((ext_vector_type(4)))  int      v4i;

#define DIN 12544
#define HID 1024

#define BM 128
#define BN 128
#define BK 32
#define LDP 40   // padded LDS row pitch (halves): 80B -> 20-bank step, conflict-free

#if defined(__has_builtin)
# if __has_builtin(__builtin_amdgcn_global_load_async_to_lds_b128)
#  define BH_ASYNC 1
# endif
#endif
#ifndef BH_ASYNC
# define BH_ASYNC 0
#endif

#define AS1 __attribute__((address_space(1)))
#define AS3 __attribute__((address_space(3)))

static __device__ __forceinline__ v16h frag_cat(v8h lo, v8h hi) {
  return __builtin_shufflevector(lo, hi, 0,1,2,3,4,5,6,7,8,9,10,11,12,13,14,15);
}

// One 16-byte (8-half) chunk: global -> LDS.
static __device__ __forceinline__ void bh_stage_chunk(const _Float16* __restrict__ g,
                                                      _Float16* l) {
#if BH_ASYNC
  // async DMA into LDS, tracked by ASYNCcnt
  __builtin_amdgcn_global_load_async_to_lds_b128((AS1 v4i*)(uintptr_t)g,
                                                 (AS3 v4i*)l, 0, 0);
#else
  *(v8h*)l = *(const v8h*)g;   // sync fallback: vmem load + ds_store
#endif
}

// ---------------------------------------------------------------------------
// fp32 -> f16 elementwise convert
// ---------------------------------------------------------------------------
__global__ __launch_bounds__(256) void bh_cvt_f16(const float* __restrict__ in,
                                                  _Float16* __restrict__ out,
                                                  long n) {
  long i = (long)blockIdx.x * blockDim.x + threadIdx.x;
  long stride = (long)gridDim.x * blockDim.x;
  for (; i < n; i += stride) out[i] = (_Float16)in[i];
}

// ---------------------------------------------------------------------------
// Tiled transpose + convert: in [R,C] fp32 -> out [C,R] f16.  R,C % 32 == 0.
// block (32,8), grid (C/32, R/32)
// ---------------------------------------------------------------------------
__global__ __launch_bounds__(256) void bh_transpose_cvt(const float* __restrict__ in,
                                                        _Float16* __restrict__ out,
                                                        int R, int C) {
  __shared__ _Float16 tile[32][33];
  int c0 = blockIdx.x * 32;
  int r0 = blockIdx.y * 32;
  for (int rr = threadIdx.y; rr < 32; rr += 8) {
    tile[rr][threadIdx.x] = (_Float16)in[(size_t)(r0 + rr) * C + (c0 + threadIdx.x)];
  }
  __syncthreads();
  for (int cc = threadIdx.y; cc < 32; cc += 8) {
    out[(size_t)(c0 + cc) * R + (r0 + threadIdx.x)] = tile[threadIdx.x][cc];
  }
}

// ---------------------------------------------------------------------------
// Fuse Wc [1024,4] and Wr [1024,12] into WheadT [16,1024] f16 (transposed)
// ---------------------------------------------------------------------------
__global__ __launch_bounds__(256) void bh_build_headw(const float* __restrict__ Wc,
                                                      const float* __restrict__ Wr,
                                                      _Float16* __restrict__ WT) {
  int idx = blockIdx.x * blockDim.x + threadIdx.x;
  if (idx >= 16 * HID) return;
  int n = idx >> 10;
  int k = idx & (HID - 1);
  float v = (n < 4) ? Wc[(size_t)k * 4 + n] : Wr[(size_t)k * 12 + (n - 4)];
  WT[(size_t)n * HID + k] = (_Float16)v;
}

// ---------------------------------------------------------------------------
// WMMA GEMM: C[M,N] = relu(A[M,K] @ B[K,N] + bias), B given as BT [N,K] f16.
// Block 256 threads = 8 waves, block tile 128x128, BK=32.
// Wave grid 4(M) x 2(N); each wave: 32x64 = 2x4 tiles of 16x16 (64 acc VGPRs).
// Tiles staged to LDS with double-buffered async global->LDS copies.
// grid = (ceil(M/128), N/128)
// ---------------------------------------------------------------------------
__global__ __launch_bounds__(256) void bh_gemm_relu(const _Float16* __restrict__ A,
                                                    const _Float16* __restrict__ BT,
                                                    const float* __restrict__ bias,
                                                    _Float16* __restrict__ C,
                                                    int M, int N, int K) {
  __shared__ _Float16 smA[2][BM * LDP];
  __shared__ _Float16 smB[2][BN * LDP];

  const int t    = threadIdx.x;
  const int lane = t & 31;
  const int wave = t >> 5;
  const int l  = lane & 15;
  const int hi = lane >> 4;

  const int blockM = blockIdx.x * BM;
  const int blockN = blockIdx.y * BN;
  const int wm = (wave & 3) * 32;   // wave row base within block tile
  const int wn = (wave >> 2) * 64;  // wave col base within block tile

  // Staging map: each tile is 128 rows x 4 b128-chunks = 512 chunks; 2/thread.
  const int c0r = t >> 2;            // rows 0..63
  const int c1r = c0r + 64;          // rows 64..127
  const int ck  = t & 3;             // which 8-half group in the 32-half row

  int ra0 = blockM + c0r; if (ra0 >= M) ra0 = M - 1;   // clamp (masked at store)
  int ra1 = blockM + c1r; if (ra1 >= M) ra1 = M - 1;
  const int rb0 = blockN + c0r;                         // N % BN == 0
  const int rb1 = blockN + c1r;

  auto stage = [&](int buf, int k0) {
    const int kc = k0 + ck * 8;
    bh_stage_chunk(A  + (size_t)ra0 * K + kc, &smA[buf][c0r * LDP + ck * 8]);
    bh_stage_chunk(A  + (size_t)ra1 * K + kc, &smA[buf][c1r * LDP + ck * 8]);
    bh_stage_chunk(BT + (size_t)rb0 * K + kc, &smB[buf][c0r * LDP + ck * 8]);
    bh_stage_chunk(BT + (size_t)rb1 * K + kc, &smB[buf][c1r * LDP + ck * 8]);
  };
  auto settle = [&]() {
#if BH_ASYNC
    asm volatile("s_wait_asynccnt 0x0" ::: "memory");   // my DMAs landed in LDS
#endif
    __syncthreads();                                    // everyone's landed / read
  };

  v8f acc[2][4];
#pragma unroll
  for (int i = 0; i < 2; ++i)
#pragma unroll
    for (int j = 0; j < 4; ++j) acc[i][j] = (v8f)(0.0f);

  stage(0, 0);
  settle();

  int buf = 0;
  for (int k0 = 0; k0 < K; k0 += BK) {
    if (k0 + BK < K) stage(buf ^ 1, k0 + BK);   // overlap DMA with WMMA below

    v16h a[2], b[4];
#pragma unroll
    for (int i = 0; i < 2; ++i) {
      // A frag: lane(l,hi) -> row wm+i*16+l ; K chunks hi*8 and 16+hi*8
      const _Float16* p = &smA[buf][(wm + i * 16 + l) * LDP + hi * 8];
      a[i] = frag_cat(*(const v8h*)p, *(const v8h*)(p + 16));
    }
#pragma unroll
    for (int j = 0; j < 4; ++j) {
      // B frag: lane(l,hi) -> col wn+j*16+l ; K chunks hi*8 and 16+hi*8
      const _Float16* p = &smB[buf][(wn + j * 16 + l) * LDP + hi * 8];
      b[j] = frag_cat(*(const v8h*)p, *(const v8h*)(p + 16));
    }
#pragma unroll
    for (int i = 0; i < 2; ++i)
#pragma unroll
      for (int j = 0; j < 4; ++j)
        acc[i][j] = __builtin_amdgcn_wmma_f32_16x16x32_f16(
            false, a[i], false, b[j], (short)0, acc[i][j], false, false);

    settle();
    buf ^= 1;
  }

  // Epilogue: bias + ReLU, store f16.
  // C/D layout: VGPR r, lane (l,hi) -> element (M = r + 8*hi, N = l)
#pragma unroll
  for (int i = 0; i < 2; ++i) {
#pragma unroll
    for (int j = 0; j < 4; ++j) {
      const int col = blockN + wn + j * 16 + l;
      const float bb = bias[col];
#pragma unroll
      for (int r = 0; r < 8; ++r) {
        const int row = blockM + wm + i * 16 + hi * 8 + r;
        if (row < M) {
          float v = acc[i][j][r] + bb;
          v = v > 0.0f ? v : 0.0f;
          C[(size_t)row * N + col] = (_Float16)v;
        }
      }
    }
  }
}

// ---------------------------------------------------------------------------
// Head GEMM: logits[M,16] (f32) = H2[M,1024] @ Whead[1024,16], WT = [16,1024].
// One wave per 16 rows.  grid = ceil(M/16), block = 32.
// ---------------------------------------------------------------------------
__global__ __launch_bounds__(32) void bh_head_gemm(const _Float16* __restrict__ H2,
                                                   const _Float16* __restrict__ WT,
                                                   float* __restrict__ logits,
                                                   int M) {
  const int lane = threadIdx.x & 31;
  const int l  = lane & 15;
  const int hi = lane >> 4;
  const int m0 = blockIdx.x * 16;

  v8f acc = (v8f)(0.0f);
  for (int k0 = 0; k0 < HID; k0 += 32) {
    int row = m0 + l;
    if (row >= M) row = M - 1;
    const _Float16* pa = H2 + (size_t)row * HID + (k0 + hi * 8);
    v16h a = frag_cat(*(const v8h*)pa, *(const v8h*)(pa + 16));
    const _Float16* pb = WT + (size_t)l * HID + (k0 + hi * 8);
    v16h b = frag_cat(*(const v8h*)pb, *(const v8h*)(pb + 16));
    acc = __builtin_amdgcn_wmma_f32_16x16x32_f16(false, a, false, b,
                                                 (short)0, acc, false, false);
  }
#pragma unroll
  for (int r = 0; r < 8; ++r) {
    const int row = m0 + hi * 8 + r;
    if (row < M) logits[(size_t)row * 16 + l] = acc[r];
  }
}

// ---------------------------------------------------------------------------
// Finish: add head biases, softmax over 4 classes, emit concatenated outputs.
// d_out = [class_probs (M x 4) ; box_pred (M x 12)]
// ---------------------------------------------------------------------------
__global__ __launch_bounds__(256) void bh_finish(const float* __restrict__ logits,
                                                 const float* __restrict__ bc,
                                                 const float* __restrict__ br,
                                                 float* __restrict__ out, int M) {
  int r = blockIdx.x * blockDim.x + threadIdx.x;
  if (r >= M) return;
  float v[16];
#pragma unroll
  for (int j = 0; j < 16; ++j)
    v[j] = logits[(size_t)r * 16 + j] + ((j < 4) ? bc[j] : br[j - 4]);

  float m = v[0];
#pragma unroll
  for (int j = 1; j < 4; ++j) m = v[j] > m ? v[j] : m;
  float e[4], s = 0.0f;
#pragma unroll
  for (int j = 0; j < 4; ++j) { e[j] = expf(v[j] - m); s += e[j]; }
  float inv = 1.0f / s;
#pragma unroll
  for (int j = 0; j < 4; ++j) out[(size_t)r * 4 + j] = e[j] * inv;

  float* box = out + (size_t)M * 4;
#pragma unroll
  for (int j = 0; j < 12; ++j) box[(size_t)r * 12 + j] = v[4 + j];
}

// ---------------------------------------------------------------------------
extern "C" void kernel_launch(void* const* d_in, const int* in_sizes, int n_in,
                              void* d_out, int out_size, void* d_ws, size_t ws_size,
                              hipStream_t stream) {
  const float* X  = (const float*)d_in[0];
  const float* W1 = (const float*)d_in[1];
  const float* b1 = (const float*)d_in[2];
  const float* W2 = (const float*)d_in[3];
  const float* b2 = (const float*)d_in[4];
  const float* Wc = (const float*)d_in[5];
  const float* bc = (const float*)d_in[6];
  const float* Wr = (const float*)d_in[7];
  const float* br = (const float*)d_in[8];

  const int M = in_sizes[0] / DIN;  // 8000

  // ---- workspace layout (256B aligned regions) ----
  char* base = (char*)d_ws;
  size_t off = 0;
  auto take = [&](size_t bytes) -> char* {
    char* p = base + off;
    off += (bytes + 255) & ~(size_t)255;
    return p;
  };
  _Float16* Xh   = (_Float16*)take((size_t)M * DIN * sizeof(_Float16));
  _Float16* W1T  = (_Float16*)take((size_t)DIN * HID * sizeof(_Float16));
  _Float16* W2T  = (_Float16*)take((size_t)HID * HID * sizeof(_Float16));
  _Float16* WHT  = (_Float16*)take((size_t)16 * HID * sizeof(_Float16));
  _Float16* H1   = (_Float16*)take((size_t)M * HID * sizeof(_Float16));
  _Float16* H2   = (_Float16*)take((size_t)M * HID * sizeof(_Float16));
  float*    LG   = (float*)take((size_t)M * 16 * sizeof(float));
  (void)ws_size; (void)n_in; (void)out_size;

  // 1) convert X to f16
  bh_cvt_f16<<<8192, 256, 0, stream>>>(X, Xh, (long)M * DIN);

  // 2) transpose+convert weights (dims are multiples of 32)
  bh_transpose_cvt<<<dim3(HID / 32, DIN / 32), dim3(32, 8), 0, stream>>>(W1, W1T, DIN, HID);
  bh_transpose_cvt<<<dim3(HID / 32, HID / 32), dim3(32, 8), 0, stream>>>(W2, W2T, HID, HID);
  bh_build_headw<<<(16 * HID + 255) / 256, 256, 0, stream>>>(Wc, Wr, WHT);

  // 3) GEMM1: H1 = relu(Xh @ W1 + b1)
  bh_gemm_relu<<<dim3((M + BM - 1) / BM, HID / BN), 256, 0, stream>>>(
      Xh, W1T, b1, H1, M, HID, DIN);
  // 4) GEMM2: H2 = relu(H1 @ W2 + b2)
  bh_gemm_relu<<<dim3((M + BM - 1) / BM, HID / BN), 256, 0, stream>>>(
      H1, W2T, b2, H2, M, HID, HID);
  // 5) head GEMM: LG = H2 @ [Wc|Wr]
  bh_head_gemm<<<(M + 15) / 16, 32, 0, stream>>>(H2, WHT, LG, M);
  // 6) bias + softmax + pack outputs
  bh_finish<<<(M + 255) / 256, 256, 0, stream>>>(LG, bc, br, (float*)d_out, M);
}
